// ReidRNSelf_91233695302213
// MI455X (gfx1250) — matrix operand order repaired and checked by reading, not compiled, not run on hardware
//
#include <hip/hip_runtime.h>

// ---------------------------------------------------------------- types
typedef __bf16 bf16_t;
typedef bf16_t v16bf __attribute__((ext_vector_type(16)));
typedef float  v8f   __attribute__((ext_vector_type(8)));

union AFrag { v16bf v; uint4 q[2]; unsigned short u[16]; };
union V8FU  { v8f v; float f[8]; };

__device__ __forceinline__ unsigned short f2bf(float f) {
  unsigned int u = __float_as_uint(f);
  unsigned int r = ((u >> 16) & 1u) + 0x7FFFu;     // round-to-nearest-even
  return (unsigned short)((u + r) >> 16);
}
__device__ __forceinline__ float bf2f(unsigned short h) {
  return __uint_as_float(((unsigned int)h) << 16);
}

__device__ __forceinline__ v8f wmma_bf16(v16bf a, v16bf b, v8f c) {
  // v_wmma_f32_16x16x32_bf16  D = A(16x32) * B(32x16) + C(16x16 f32)
  return __builtin_amdgcn_wmma_f32_16x16x32_bf16(
      false, a, false, b, (short)0, c, false, false);
}

// CDNA5 async copy: global -> LDS, ASYNCcnt-tracked (ISA 08_async_tensor §4).
// lds_byte = wave-relative LDS byte address (low 32 bits of generic pointer).
__device__ __forceinline__ void async_ld_lds_b128(unsigned int lds_byte,
                                                  const void* g) {
  asm volatile("global_load_async_to_lds_b128 %0, %1, off"
               :: "v"(lds_byte), "v"(g) : "memory");
}
__device__ __forceinline__ void wait_async_zero() {
  asm volatile("s_wait_asynccnt 0x0" ::: "memory");
}

// A fragment (16x32 bf16) from a row-major [rows][ld] bf16 buffer.
// ISA layout: lanes 0-15 -> M=lane, K = k0+{0..7,16..23};
//             lanes 16-31 -> M=lane-16, K = k0+{8..15,24..31}.
__device__ __forceinline__ v16bf load_a_rm(const unsigned short* src, int ld,
                                           int row0, int k0, int lane) {
  AFrag f;
  const unsigned short* p =
      src + (size_t)(row0 + (lane & 15)) * ld + k0 + ((lane >> 4) << 3);
  f.q[0] = *(const uint4*)(p);
  f.q[1] = *(const uint4*)(p + 16);
  return f.v;
}

// B fragment (32x16 bf16) from pre-packed weights: 32 bytes per lane,
// record index ((ntile*Ktiles + kt)*32 + lane).
__device__ __forceinline__ v16bf load_b_pk(const unsigned short* pB, int Ktiles,
                                           int kt, int ntile, int lane) {
  AFrag f;
  const unsigned short* p =
      pB + ((((size_t)ntile * Ktiles + kt) * 32 + lane) << 4);
  f.q[0] = *(const uint4*)(p);
  f.q[1] = *(const uint4*)(p + 8);
  return f.v;
}

// ------------------------------------------------------------- prep kernels
// xbf[(b*32+p)*2048 + c] = bf16(x[b,0,c,h,w]),  p = h*4+w
__global__ void k_conv_x(const float* __restrict__ x,
                         unsigned short* __restrict__ xbf) {
  const long total = 4096L * 2048L;
  for (long idx = (long)blockIdx.x * blockDim.x + threadIdx.x; idx < total;
       idx += (long)gridDim.x * blockDim.x) {
    int c   = (int)(idx & 2047);
    int row = (int)(idx >> 11);
    int b = row >> 5, p = row & 31;
    xbf[idx] = f2bf(x[((size_t)b * 2048 + c) * 32 + p]);
  }
}

// Generic B-packer: W row-major [N][K] fp32 -> bf16 B fragments, Npad tiles.
__global__ void k_pack_b(const float* __restrict__ W,
                         unsigned short* __restrict__ dst, int N, int K,
                         int Ntiles) {
  const int Ktiles = K >> 5;
  const long total = (long)Ntiles * Ktiles * 512;
  for (long idx = (long)blockIdx.x * blockDim.x + threadIdx.x; idx < total;
       idx += (long)gridDim.x * blockDim.x) {
    int j = (int)(idx & 15);
    int lane = (int)((idx >> 4) & 31);
    long t = idx >> 9;
    int kt = (int)(t % Ktiles), ntile = (int)(t / Ktiles);
    int k = kt * 32 + ((lane >> 4) << 3) + (j & 7) + ((j >> 3) << 4);
    int n = ntile * 16 + (lane & 15);
    float v = (n < N) ? W[(size_t)n * K + k] : 0.f;
    dst[idx] = f2bf(v);
  }
}

// Packed B for the fused u|v|conv GEMM: N logical = [Wp(512)|Wa(512)|conv(32)],
// padded to 1152 (72 tiles), K = 2048 (64 tiles).
__global__ void k_pack_uv(const float* __restrict__ g1_w,
                          const float* __restrict__ conv_w,
                          unsigned short* __restrict__ dst) {
  const int Ktiles = 64, Ntiles = 72;
  const long total = (long)Ntiles * Ktiles * 512;
  for (long idx = (long)blockIdx.x * blockDim.x + threadIdx.x; idx < total;
       idx += (long)gridDim.x * blockDim.x) {
    int j = (int)(idx & 15);
    int lane = (int)((idx >> 4) & 31);
    long t = idx >> 9;
    int kt = (int)(t % Ktiles), ntile = (int)(t / Ktiles);
    int k = kt * 32 + ((lane >> 4) << 3) + (j & 7) + ((j >> 3) << 4);
    int n = ntile * 16 + (lane & 15);
    float v = 0.f;
    if (n < 512)        v = g1_w[(size_t)n * 4096 + k];                // Wp
    else if (n < 1024)  v = g1_w[(size_t)(n - 512) * 4096 + 2048 + k]; // Wa
    else if (n < 1056)  v = conv_w[(size_t)(n - 1024) * 2048 + k];
    dst[idx] = f2bf(v);
  }
}

__global__ void k_zero_f32(float* p, long n) {
  for (long i = (long)blockIdx.x * blockDim.x + threadIdx.x; i < n;
       i += (long)gridDim.x * blockDim.x)
    p[i] = 0.f;
}

// ---------------------------------------------- GEMM #1: u, v, conv (fused)
// C(4096 x 1152) = xbf(4096 x 2048) * Bcat.  Wave tile 16x64 (4 accumulators
// per A fragment); register double-buffering so iteration kt+1's 10 loads are
// issued before kt's 4 WMMAs (hides load latency behind the XDL pipe).
__global__ __launch_bounds__(256) void k_gemm_uv(
    const unsigned short* __restrict__ xbf, const unsigned short* __restrict__ pB,
    const float* __restrict__ g1_b, unsigned short* __restrict__ ubf,
    unsigned short* __restrict__ vbf, float* __restrict__ rawg) {
  const int lane = threadIdx.x & 31;
  const int wv = threadIdx.x >> 5;               // 8 waves: 4 M-tiles x 2 N-halves
  const int m0 = blockIdx.x * 64 + (wv & 3) * 16;
  const int n0 = blockIdx.y * 128 + (wv >> 2) * 64;
  const int nt0 = n0 >> 4;
  const v8f vz = {0.f, 0.f, 0.f, 0.f, 0.f, 0.f, 0.f, 0.f};
  v8f acc0 = vz, acc1 = vz, acc2 = vz, acc3 = vz;

  v16bf a  = load_a_rm(xbf, 2048, m0, 0, lane);
  v16bf b0 = load_b_pk(pB, 64, 0, nt0 + 0, lane);
  v16bf b1 = load_b_pk(pB, 64, 0, nt0 + 1, lane);
  v16bf b2 = load_b_pk(pB, 64, 0, nt0 + 2, lane);
  v16bf b3 = load_b_pk(pB, 64, 0, nt0 + 3, lane);

  for (int kt = 1; kt < 64; ++kt) {
    v16bf an = load_a_rm(xbf, 2048, m0, kt * 32, lane);
    v16bf c0 = load_b_pk(pB, 64, kt, nt0 + 0, lane);
    v16bf c1 = load_b_pk(pB, 64, kt, nt0 + 1, lane);
    v16bf c2 = load_b_pk(pB, 64, kt, nt0 + 2, lane);
    v16bf c3 = load_b_pk(pB, 64, kt, nt0 + 3, lane);
    acc0 = wmma_bf16(a, b0, acc0);
    acc1 = wmma_bf16(a, b1, acc1);
    acc2 = wmma_bf16(a, b2, acc2);
    acc3 = wmma_bf16(a, b3, acc3);
    a = an; b0 = c0; b1 = c1; b2 = c2; b3 = c3;
  }
  acc0 = wmma_bf16(a, b0, acc0);
  acc1 = wmma_bf16(a, b1, acc1);
  acc2 = wmma_bf16(a, b2, acc2);
  acc3 = wmma_bf16(a, b3, acc3);

  // D layout: VGPR j -> lanes0-15:(M=j,N=lane); lanes16-31:(M=8+j,N=lane-16)
  const int colb = lane & 15;
  const int rowbase = m0 + ((lane >> 4) << 3);
  V8FU t[4];
  t[0].v = acc0; t[1].v = acc1; t[2].v = acc2; t[3].v = acc3;
  for (int tt = 0; tt < 4; ++tt) {
    const int col = n0 + tt * 16 + colb;
    for (int j = 0; j < 8; ++j) {
      const int row = rowbase + j;
      const float v = t[tt].f[j];
      if (col < 512)
        ubf[(size_t)row * 512 + col] = f2bf(v + g1_b[col]);      // fold g1_b
      else if (col < 1024)
        vbf[(size_t)row * 512 + (col - 512)] = f2bf(v);
      else if (col < 1056)
        rawg[(size_t)row * 32 + (col - 1024)] = v;
    }
  }
}

// 2x2 avg-pool + conv bias: rawg[b][p=h*4+w][o] -> glbl[b][o*8 + ph*2 + pw]
__global__ void k_pool(const float* __restrict__ rawg,
                       const float* __restrict__ conv_b,
                       float* __restrict__ glbl) {
  int i = blockIdx.x * blockDim.x + threadIdx.x;   // 128*256
  if (i >= 128 * 256) return;
  int b = i >> 8, r = i & 255;
  int o = r >> 3, ph = (r >> 1) & 3, pw = r & 1;
  float s = 0.f;
  for (int dh = 0; dh < 2; ++dh)
    for (int dw = 0; dw < 2; ++dw) {
      int p = (2 * ph + dh) * 4 + (2 * pw + dw);
      s += rawg[((size_t)b * 32 + p) * 32 + o];
    }
  glbl[i] = s * 0.25f + conv_b[o];
}

// ----------------------------------- fused pairwise MLP (dominant 69 GFLOP)
// block = 4 waves; wave w handles a = blk.a0 + w, both p-halves.
// The 16 u-rows of the current p-half (shared by all 4 waves) are staged into
// LDS once per block via CDNA5 async global->LDS copies (ASYNCcnt). h rows are
// then built with packed bf16 adds + sign-bit relu; layers 2/3/4 via WMMA with
// per-wave LDS re-staging for the D->A transpose; row sums via ds_add_f32
// then one global atomic per column.
__global__ __launch_bounds__(128) void k_pair_mlp(
    const unsigned short* __restrict__ ubf, const unsigned short* __restrict__ vbf,
    const unsigned short* __restrict__ pg2, const unsigned short* __restrict__ pg3,
    const unsigned short* __restrict__ pg4, const float* __restrict__ g2_b,
    const float* __restrict__ g3_b, const float* __restrict__ g4_b,
    float* __restrict__ x_g) {
  __shared__ __align__(16) unsigned short h2buf[4][16 * 256];  // 32 KB
  __shared__ __align__(16) unsigned short ubuf[16 * 512];      // 16 KB
  __shared__ float sacc[256];

  const int tid = threadIdx.x;
  const int lane = tid & 31;
  const int w = tid >> 5;
  const int b = blockIdx.x >> 3;
  const int a = (blockIdx.x & 7) * 4 + w;

  for (int i = tid; i < 256; i += 128) sacc[i] = 0.f;

  const unsigned short* urows = ubf + (size_t)b * 32 * 512;
  const unsigned short* vrow  = vbf + ((size_t)b * 32 + a) * 512;
  unsigned short* hl = h2buf[w];
  const unsigned int ubase = (unsigned int)(size_t)(void*)ubuf;
  const int colb = lane & 15;
  const int rowb = (lane >> 4) << 3;
  const int koff0 = (lane >> 4) << 3;

  for (int ph = 0; ph < 2; ++ph) {
    const int p0 = ph * 16;

    // ---- stage u rows p0..p0+15 (16 KB contiguous) into LDS, async path
    __syncthreads();                        // previous p-half done with ubuf
    {
      const char* usrc = (const char*)(urows + (size_t)p0 * 512);
      for (int i = tid; i < 1024; i += 128)
        async_ld_lds_b128(ubase + (unsigned)i * 16u, usrc + (size_t)i * 16);
      wait_async_zero();                    // this wave's copies landed
      __syncthreads();                      // everyone's copies visible
    }

    v8f acc[16];
    const v8f vz = {0.f, 0.f, 0.f, 0.f, 0.f, 0.f, 0.f, 0.f};
#pragma unroll
    for (int nt = 0; nt < 16; ++nt) acc[nt] = vz;

    // ---- layer 2: h(16x512) @ g2^T -> 16x256, K tiles = 16
    for (int kt = 0; kt < 16; ++kt) {
      const int k0 = kt * 32;
      const unsigned short* up =
          ubuf + (size_t)(lane & 15) * 512 + k0 + koff0;         // ds_load
      const unsigned short* vp = vrow + k0 + koff0;              // global
      AFrag ua, va, ha;
      ua.q[0] = *(const uint4*)(up);
      ua.q[1] = *(const uint4*)(up + 16);
      va.q[0] = *(const uint4*)(vp);
      va.q[1] = *(const uint4*)(vp + 16);
      ha.v = ua.v + va.v;                    // packed bf16 add (u' + v)
#pragma unroll
      for (int j = 0; j < 16; ++j)           // relu via sign bit (-0 -> +0)
        ha.u[j] = (ha.u[j] & 0x8000u) ? (unsigned short)0 : ha.u[j];
#pragma unroll
      for (int nt = 0; nt < 16; ++nt)
        acc[nt] = wmma_bf16(ha.v, load_b_pk(pg2, 16, kt, nt, lane), acc[nt]);
    }
    // relu(+g2_b) -> per-wave LDS (row-major 16x256 bf16)
#pragma unroll
    for (int nt = 0; nt < 16; ++nt) {
      const int col = nt * 16 + colb;
      const float bias = g2_b[col];
      V8FU t; t.v = acc[nt];
#pragma unroll
      for (int j = 0; j < 8; ++j) {
        float v = t.f[j] + bias;
        hl[(rowb + j) * 256 + col] = f2bf(v > 0.f ? v : 0.f);
      }
    }

    // ---- layer 3: 16x256 @ g3^T, K tiles = 8
#pragma unroll
    for (int nt = 0; nt < 16; ++nt) acc[nt] = vz;
    for (int kt = 0; kt < 8; ++kt) {
      v16bf afr = load_a_rm(hl, 256, 0, kt * 32, lane);
#pragma unroll
      for (int nt = 0; nt < 16; ++nt)
        acc[nt] = wmma_bf16(afr, load_b_pk(pg3, 8, kt, nt, lane), acc[nt]);
    }
#pragma unroll
    for (int nt = 0; nt < 16; ++nt) {
      const int col = nt * 16 + colb;
      const float bias = g3_b[col];
      V8FU t; t.v = acc[nt];
#pragma unroll
      for (int j = 0; j < 8; ++j) {
        float v = t.f[j] + bias;
        hl[(rowb + j) * 256 + col] = f2bf(v > 0.f ? v : 0.f);
      }
    }

    // ---- layer 4: 16x256 @ g4^T, then relu(+g4_b) and row-sum
#pragma unroll
    for (int nt = 0; nt < 16; ++nt) acc[nt] = vz;
    for (int kt = 0; kt < 8; ++kt) {
      v16bf afr = load_a_rm(hl, 256, 0, kt * 32, lane);
#pragma unroll
      for (int nt = 0; nt < 16; ++nt)
        acc[nt] = wmma_bf16(afr, load_b_pk(pg4, 8, kt, nt, lane), acc[nt]);
    }
#pragma unroll
    for (int nt = 0; nt < 16; ++nt) {
      const int col = nt * 16 + colb;
      const float bias = g4_b[col];
      V8FU t; t.v = acc[nt];
      float part = 0.f;
#pragma unroll
      for (int j = 0; j < 8; ++j) {
        float v = t.f[j] + bias;
        part += (v > 0.f ? v : 0.f);
      }
      atomicAdd(&sacc[col], part);                    // ds_add_f32
    }
  }
  __syncthreads();
  for (int i = tid; i < 256; i += 128)
    atomicAdd(&x_g[(size_t)b * 256 + i], sacc[i]);
}

// -------------------------------------------------- small fp32 tail kernels
__global__ __launch_bounds__(256) void k_final1(
    const float* __restrict__ x_g, const float* __restrict__ glbl,
    const float* __restrict__ f1_w, const float* __restrict__ f1_b,
    const float* __restrict__ fc2_w, const float* __restrict__ fc2_b,
    float* __restrict__ y2) {
  __shared__ float sx[256], sg[256], sxf[256];
  const int b = blockIdx.x, n = threadIdx.x;
  sx[n] = x_g[(size_t)b * 256 + n];
  sg[n] = glbl[(size_t)b * 256 + n];
  __syncthreads();
  float acc = f1_b[n];
  for (int i = 0; i < 256; ++i) acc += sx[i] * f1_w[(size_t)n * 256 + i];
  sxf[n] = acc > 0.f ? acc : 0.f;
  __syncthreads();
  float a2 = fc2_b[n];
  const float* wr = fc2_w + (size_t)n * 512;
  for (int i = 0; i < 256; ++i) a2 += sxf[i] * wr[i];
  for (int i = 0; i < 256; ++i) a2 += sg[i] * wr[256 + i];
  y2[(size_t)b * 256 + n] = a2;
}

__global__ void k_bnstats(const float* __restrict__ y, int rows, int cols,
                          float* __restrict__ mu, float* __restrict__ rstd) {
  int n = threadIdx.x;
  if (n >= cols) return;
  float s = 0.f, s2 = 0.f;
  for (int r = 0; r < rows; ++r) {
    float v = y[(size_t)r * cols + n];
    s += v;
    s2 += v * v;
  }
  float m = s / rows;
  float var = s2 / rows - m * m;
  mu[n] = m;
  rstd[n] = rsqrtf(var + 1e-5f);
}

__global__ __launch_bounds__(128) void k_fc3(
    const float* __restrict__ y2, const float* __restrict__ mu2,
    const float* __restrict__ rstd2, const float* __restrict__ bn2_g,
    const float* __restrict__ bn2_b, const float* __restrict__ fc3_w,
    const float* __restrict__ fc3_b, float* __restrict__ y3) {
  __shared__ float sy[256];
  const int b = blockIdx.x, m = threadIdx.x;
  for (int i = m; i < 256; i += 128) {
    float v = (y2[(size_t)b * 256 + i] - mu2[i]) * rstd2[i] * bn2_g[i] + bn2_b[i];
    sy[i] = v > 0.f ? v : 0.f;
  }
  __syncthreads();
  float acc = fc3_b[m];
  for (int i = 0; i < 256; ++i) acc += sy[i] * fc3_w[(size_t)m * 256 + i];
  y3[(size_t)b * 128 + m] = acc;
}

__global__ __launch_bounds__(128) void k_out(
    const float* __restrict__ y3, const float* __restrict__ mu3,
    const float* __restrict__ rstd3, const float* __restrict__ bn3_g,
    const float* __restrict__ bn3_b, float* __restrict__ out) {
  const int b = blockIdx.x, m = threadIdx.x;
  float v = (y3[(size_t)b * 128 + m] - mu3[m]) * rstd3[m] * bn3_g[m] + bn3_b[m];
  out[(size_t)b * 128 + m] = v > 0.f ? v : 0.f;
}

// ------------------------------------------------------------------ launch
extern "C" void kernel_launch(void* const* d_in, const int* in_sizes, int n_in,
                              void* d_out, int out_size, void* d_ws,
                              size_t ws_size, hipStream_t stream) {
  (void)in_sizes; (void)n_in; (void)out_size; (void)ws_size;
  const float* x      = (const float*)d_in[0];
  const float* conv_w = (const float*)d_in[1];
  const float* conv_b = (const float*)d_in[2];
  const float* g1_w   = (const float*)d_in[3];
  const float* g1_b   = (const float*)d_in[4];
  const float* g2_w   = (const float*)d_in[5];
  const float* g2_b   = (const float*)d_in[6];
  const float* g3_w   = (const float*)d_in[7];
  const float* g3_b   = (const float*)d_in[8];
  const float* g4_w   = (const float*)d_in[9];
  const float* g4_b   = (const float*)d_in[10];
  const float* f1_w   = (const float*)d_in[11];
  const float* f1_b   = (const float*)d_in[12];
  const float* fc2_w  = (const float*)d_in[13];
  const float* fc2_b  = (const float*)d_in[14];
  const float* bn2_g  = (const float*)d_in[15];
  const float* bn2_b  = (const float*)d_in[16];
  const float* fc3_w  = (const float*)d_in[17];
  const float* fc3_b  = (const float*)d_in[18];
  const float* bn3_g  = (const float*)d_in[19];
  const float* bn3_b  = (const float*)d_in[20];

  char* w = (char*)d_ws;
  unsigned short* xbf   = (unsigned short*)(w + 0);           // 16,777,216
  unsigned short* pB_uv = (unsigned short*)(w + 16777216);    //  4,718,592
  unsigned short* pB_g2 = (unsigned short*)(w + 21495808);    //    262,144
  unsigned short* pB_g3 = (unsigned short*)(w + 21757952);    //    131,072
  unsigned short* pB_g4 = (unsigned short*)(w + 21889024);    //    131,072
  unsigned short* ubf   = (unsigned short*)(w + 22020096);    //  4,194,304
  unsigned short* vbf   = (unsigned short*)(w + 26214400);    //  4,194,304
  float* rawg  = (float*)(w + 30408704);                      //    524,288
  float* glbl  = (float*)(w + 30932992);                      //    131,072
  float* x_g   = (float*)(w + 31064064);                      //    131,072
  float* y2    = (float*)(w + 31195136);                      //    131,072
  float* mu2   = (float*)(w + 31326208);                      //      1,024
  float* rstd2 = (float*)(w + 31327232);                      //      1,024
  float* y3    = (float*)(w + 31328256);                      //     65,536
  float* mu3   = (float*)(w + 31393792);                      //        512
  float* rstd3 = (float*)(w + 31394304);                      //        512

  k_conv_x<<<2048, 256, 0, stream>>>(x, xbf);
  k_pack_uv<<<4096, 256, 0, stream>>>(g1_w, conv_w, pB_uv);
  k_pack_b<<<512, 256, 0, stream>>>(g2_w, pB_g2, 256, 512, 16);
  k_pack_b<<<256, 256, 0, stream>>>(g3_w, pB_g3, 256, 256, 16);
  k_pack_b<<<256, 256, 0, stream>>>(g4_w, pB_g4, 256, 256, 16);
  k_zero_f32<<<128, 256, 0, stream>>>(x_g, 128L * 256L);

  k_gemm_uv<<<dim3(64, 9), 256, 0, stream>>>(xbf, pB_uv, g1_b, ubf, vbf, rawg);
  k_pool<<<128, 256, 0, stream>>>(rawg, conv_b, glbl);

  k_pair_mlp<<<1024, 128, 0, stream>>>(ubf, vbf, pB_g2, pB_g3, pB_g4, g2_b,
                                       g3_b, g4_b, x_g);

  k_final1<<<128, 256, 0, stream>>>(x_g, glbl, f1_w, f1_b, fc2_w, fc2_b, y2);
  k_bnstats<<<1, 256, 0, stream>>>(y2, 128, 256, mu2, rstd2);
  k_fc3<<<128, 128, 0, stream>>>(y2, mu2, rstd2, bn2_g, bn2_b, fc3_w, fc3_b, y3);
  k_bnstats<<<1, 128, 0, stream>>>(y3, 128, 128, mu3, rstd3);
  k_out<<<128, 128, 0, stream>>>(y3, mu3, rstd3, bn3_g, bn3_b, (float*)d_out);
}
